// JointEdgeSegLoss_OHEM_85607288144420
// MI455X (gfx1250) — compile-verified, matrix operand order checked by reading, not run on hardware
//
#include <hip/hip_runtime.h>
#include <stdint.h>

// ---------------- problem constants ----------------
#define NUM_C   19
#define IGN     255
#define HW      262144          // 512*512 = 2^18
#define NB      4
#define NTOT    (NB * HW)       // 1,048,576
#define MINKEPT 100000u
#define BLOCKS_A 512
#define BLOCKS_H 256
#define BLOCKS_C 256

typedef _Float16 half16 __attribute__((ext_vector_type(16)));
typedef float    float8 __attribute__((ext_vector_type(8)));

// ---------------- WMMA block reduction (blockDim == 256) ----------------
// Sums 256 f32 values exactly-ish via two chained v_wmma_f32_16x16x32_f16:
// value matrix (16x16, zero-padded to 16x32) x ones(32x16). Each value is
// split v = hi + lo in f16; products with 1.0 are exact, accumulate in f32.
// Every wave redundantly computes the same total -> EXEC is all-ones for
// every WMMA (ISA requirement) and the return value is uniform.
__device__ __forceinline__ float block_reduce_256(float v, float* buf) {
  const int tid = threadIdx.x;
  buf[tid] = v;
  __syncthreads();
  const int lane = tid & 31;
  const int m    = lane & 15;
  const int kb   = (lane < 16) ? 0 : 8;   // A f16 layout: lanes 0-15 K=0..7, lanes 16-31 K=8..15
  half16 ahi = {};
  half16 alo = {};
  half16 bones;
#pragma unroll
  for (int j = 0; j < 16; ++j) bones[j] = (_Float16)1.0f;
#pragma unroll
  for (int j = 0; j < 8; ++j) {
    float x = buf[m * 16 + kb + j];
    _Float16 h = (_Float16)x;
    ahi[j] = h;
    alo[j] = (_Float16)(x - (float)h);
  }
  float8 d = {};
  d = __builtin_amdgcn_wmma_f32_16x16x32_f16(false, alo, false, bones, (short)0, d, false, false);
  d = __builtin_amdgcn_wmma_f32_16x16x32_f16(false, ahi, false, bones, (short)0, d, false, false);
  // D layout: lanes 0-15 hold rows 0..7 across the 8 VGPRs, lanes 16-31 rows 8..15.
  float part = 0.f;
#pragma unroll
  for (int j = 0; j < 8; ++j) part += d[j];
  part += __shfl_xor(part, 16, 32);
  __syncthreads();               // buf may be reused by caller
  return part;
}

// ---------------- init: zero the atomic accumulators ----------------
__global__ void zero_init(unsigned* hist, unsigned* attCnt, unsigned long long* attSum) {
  int t = threadIdx.x;
  for (int i = t; i < 4 * 256; i += 256) hist[i] = 0u;
  if (t < NB * NUM_C) { attCnt[t] = 0u; attSum[t] = 0ull; }
}

// ---------------- Pass A: single fused pass over segin ----------------
// Computes per-pixel log-softmax stats, stores nll + mask_prob bits,
// level-0 selection histogram, BCE partial sums, attention per-(img,class)
// {count, fixed-point nll-sum} accumulators.
__global__ void __launch_bounds__(256) pass_a(
    const float* __restrict__ segin, const float* __restrict__ edgein,
    const int* __restrict__ segmask, const float* __restrict__ edgemask,
    unsigned* __restrict__ mp, float* __restrict__ nllb,
    unsigned* __restrict__ hist0, unsigned* __restrict__ attCnt,
    unsigned long long* __restrict__ attSum, float* __restrict__ pBce) {
  __shared__ unsigned           sh_hist[256];
  __shared__ unsigned           sh_attCnt[NB * NUM_C];
  __shared__ unsigned long long sh_attSum[NB * NUM_C];
  __shared__ float              sh_red[256];
  const int tid = threadIdx.x;
  for (int j = tid; j < 256; j += 256) sh_hist[j] = 0u;
  if (tid < NB * NUM_C) { sh_attCnt[tid] = 0u; sh_attSum[tid] = 0ull; }
  __syncthreads();

  float bcePos = 0.f, bceNeg = 0.f, cntPos = 0.f, cntNeg = 0.f;
  for (int i = blockIdx.x * 256 + tid; i < NTOT; i += gridDim.x * 256) {
    const int b   = i >> 18;
    const int pix = i & (HW - 1);
    const float* lp = segin + (size_t)b * NUM_C * HW + pix;
    const int t  = segmask[i];
    const bool valid = (t != IGN);
    const int tt = valid ? t : 0;

    float l[NUM_C];
    float mx = -3.4e38f;
#pragma unroll
    for (int c = 0; c < NUM_C; ++c) { l[c] = lp[(size_t)c * HW]; mx = fmaxf(mx, l[c]); }
    float s = 0.f, lt = l[0];
#pragma unroll
    for (int c = 0; c < NUM_C; ++c) {
      s += __expf(l[c] - mx);
      if (c > 0) lt = (tt == c) ? l[c] : lt;
    }
    const float lse = mx + __logf(s);
    const float nv  = lse - lt;                       // NLL at target (>= 0)
    const float p   = __expf(-nv);                    // prob at target
    const float mpv = valid ? p : 1.0f;               // invalid -> 1.0 (never kept)
    const unsigned bits = __float_as_uint(mpv);       // non-negative -> monotone bits
    mp[i]   = bits;
    nllb[i] = nv;
    atomicAdd(&sh_hist[bits >> 24], 1u);

    const float e = edgein[i];
    if (valid && e > 0.8f) {
      const int idx = b * NUM_C + tt;
      atomicAdd(&sh_attCnt[idx], 1u);
      atomicAdd(&sh_attSum[idx], (unsigned long long)(nv * 1048576.0f + 0.5f)); // 2^20 fixed point
    }
    const float tm  = edgemask[i];
    const float bce = fmaxf(e, 0.f) - e * tm + log1pf(__expf(-fabsf(e)));
    if (tm == 1.0f)      { bcePos += bce; cntPos += 1.f; }
    else if (tm == 0.0f) { bceNeg += bce; cntNeg += 1.f; }
  }
  __syncthreads();
  atomicAdd(&hist0[tid], sh_hist[tid]);
  if (tid < NB * NUM_C) {
    atomicAdd(&attCnt[tid], sh_attCnt[tid]);
    atomicAdd(&attSum[tid], sh_attSum[tid]);
  }
  float r;
  r = block_reduce_256(bcePos, sh_red); if (tid == 0) pBce[0 * BLOCKS_A + blockIdx.x] = r;
  r = block_reduce_256(bceNeg, sh_red); if (tid == 0) pBce[1 * BLOCKS_A + blockIdx.x] = r;
  r = block_reduce_256(cntPos, sh_red); if (tid == 0) pBce[2 * BLOCKS_A + blockIdx.x] = r;
  r = block_reduce_256(cntNeg, sh_red); if (tid == 0) pBce[3 * BLOCKS_A + blockIdx.x] = r;
}

// ---------------- radix-select refinement histogram (levels 1..3) ----------------
__global__ void __launch_bounds__(256) hist_level(
    const unsigned* __restrict__ mp, const unsigned* __restrict__ sel,
    unsigned* __restrict__ hist, int level) {
  __shared__ unsigned sh[256];
  sh[threadIdx.x] = 0u;
  __syncthreads();
  const unsigned prefix = sel[0];
  const int mshift = 32 - 8 * level;
  for (int i = blockIdx.x * 256 + threadIdx.x; i < NTOT; i += gridDim.x * 256) {
    const unsigned bits = mp[i];
    if ((bits >> mshift) == prefix)
      atomicAdd(&sh[(bits >> (mshift - 8)) & 0xFFu], 1u);
  }
  __syncthreads();
  atomicAdd(&hist[threadIdx.x], sh[threadIdx.x]);
}

// ---------------- scan one 256-bin level, update {prefix, rank} ----------------
__global__ void scan_level(const unsigned* __restrict__ histBase,
                           unsigned* __restrict__ sel, int level,
                           float* __restrict__ thresh) {
  unsigned prefix, rank;
  if (level == 0) { prefix = 0u; rank = MINKEPT; }
  else            { prefix = sel[0]; rank = sel[1]; }
  const unsigned* h = histBase + level * 256;
  unsigned cum = 0u; int b = 255;
  for (int i = 0; i < 256; ++i) {
    const unsigned c2 = cum + h[i];
    if (c2 >= rank) { b = i; break; }
    cum = c2;
  }
  prefix = (prefix << 8) | (unsigned)b;
  rank  -= cum;
  sel[0] = prefix; sel[1] = rank;
  if (level == 3) {
    const float kth = __uint_as_float(prefix);
    *thresh = fmaxf(0.7f, kth);
  }
}

// ---------------- Pass C: kept-pixel NLL reduction ----------------
__global__ void __launch_bounds__(256) pass_c(
    const unsigned* __restrict__ mp, const float* __restrict__ nllb,
    const int* __restrict__ segmask, const float* __restrict__ thresh,
    float* __restrict__ pC) {
  __shared__ float sh_red[256];
  const float th = *thresh;
  float cnt = 0.f, sum = 0.f;
  for (int i = blockIdx.x * 256 + threadIdx.x; i < NTOT; i += gridDim.x * 256) {
    const bool valid = (segmask[i] != IGN);
    const float mpv = __uint_as_float(mp[i]);
    if (valid && mpv <= th) { cnt += 1.f; sum += nllb[i]; }
  }
  float r;
  r = block_reduce_256(sum, sh_red); if (threadIdx.x == 0) pC[blockIdx.x] = r;
  r = block_reduce_256(cnt, sh_red); if (threadIdx.x == 0) pC[BLOCKS_C + blockIdx.x] = r;
}

// ---------------- finalize: deterministic serial combine ----------------
__global__ void finalize(const float* __restrict__ pBce, const float* __restrict__ pC,
                         const unsigned* __restrict__ attCnt,
                         const unsigned long long* __restrict__ attSum,
                         float* __restrict__ out) {
  float bcePos = 0.f, bceNeg = 0.f, cp = 0.f, cn = 0.f;
  for (int i = 0; i < BLOCKS_A; ++i) {
    bcePos += pBce[0 * BLOCKS_A + i];
    bceNeg += pBce[1 * BLOCKS_A + i];
    cp     += pBce[2 * BLOCKS_A + i];
    cn     += pBce[3 * BLOCKS_A + i];
  }
  float nsum = 0.f, ncnt = 0.f;
  for (int i = 0; i < BLOCKS_C; ++i) { nsum += pC[i]; ncnt += pC[BLOCKS_C + i]; }
  const float seg  = nsum / fmaxf(ncnt, 1.0f);                       // OHEM CE
  const float s    = cp + cn;
  const float edge = ((cn / s) * bcePos + (cp / s) * bceNeg) / (float)NTOT; // weighted BCE mean
  float att = 0.f;
  for (int b = 0; b < NB; ++b) {
    float tot = 0.f;
    for (int c = 0; c < NUM_C; ++c) tot += (float)attCnt[b * NUM_C + c];
    float wnum = 0.f, wden = 0.f;
    for (int c = 0; c < NUM_C; ++c) {
      const float binc = (float)attCnt[b * NUM_C + c];
      const float w = ((binc != 0.f) ? 1.0f : 0.0f) * (1.0f - binc / tot) + 1.0f; // UPPER_BOUND=1
      const float S = (float)((double)attSum[b * NUM_C + c] * (1.0 / 1048576.0));
      wnum += w * S;
      wden += w * binc;
    }
    att += wnum / wden;
  }
  out[0] = 1.0f * seg + 0.3f * edge + 0.1f * att;
}

// ---------------- host-side launch ----------------
extern "C" void kernel_launch(void* const* d_in, const int* in_sizes, int n_in,
                              void* d_out, int out_size, void* d_ws, size_t ws_size,
                              hipStream_t stream) {
  (void)in_sizes; (void)n_in; (void)out_size; (void)ws_size;
  const float* segin    = (const float*)d_in[0];
  const float* edgein   = (const float*)d_in[1];
  const int*   segmask  = (const int*)d_in[2];
  const float* edgemask = (const float*)d_in[3];

  char* ws = (char*)d_ws;
  unsigned* mp   = (unsigned*)ws;                                  // N u32  (4 MB)
  float*    nllb = (float*)(ws + (size_t)NTOT * 4);                // N f32  (4 MB)
  char* p = ws + (size_t)NTOT * 8;
  unsigned* hist = (unsigned*)p;              p += 4 * 256 * sizeof(unsigned);
  unsigned* attCnt = (unsigned*)p;            p += NB * NUM_C * sizeof(unsigned);
  p = (char*)(((uintptr_t)p + 7) & ~(uintptr_t)7);
  unsigned long long* attSum = (unsigned long long*)p; p += NB * NUM_C * sizeof(unsigned long long);
  unsigned* sel = (unsigned*)p;               p += 2 * sizeof(unsigned);
  float* thresh = (float*)p;                  p += 2 * sizeof(float); // keep 8B alignment
  float* pBce = (float*)p;                    p += 4 * BLOCKS_A * sizeof(float);
  float* pC   = (float*)p;                    p += 2 * BLOCKS_C * sizeof(float);

  zero_init<<<1, 256, 0, stream>>>(hist, attCnt, attSum);
  pass_a<<<BLOCKS_A, 256, 0, stream>>>(segin, edgein, segmask, edgemask,
                                       mp, nllb, hist, attCnt, attSum, pBce);
  scan_level<<<1, 1, 0, stream>>>(hist, sel, 0, thresh);
  for (int lv = 1; lv <= 3; ++lv) {
    hist_level<<<BLOCKS_H, 256, 0, stream>>>(mp, sel, hist + lv * 256, lv);
    scan_level<<<1, 1, 0, stream>>>(hist, sel, lv, thresh);
  }
  pass_c<<<BLOCKS_C, 256, 0, stream>>>(mp, nllb, segmask, thresh, pC);
  finalize<<<1, 1, 0, stream>>>(pBce, pC, attCnt, attSum, (float*)d_out);
}
